// TransformerPlant_88536455840152
// MI455X (gfx1250) — matrix operand order repaired and checked by reading, not compiled, not run on hardware
//
#include <hip/hip_runtime.h>
#include <hip/hip_bf16.h>

// ---------------- model dimensions ----------------
#define B_    16
#define T_    256
#define D_    512
#define H_    8
#define DH_   64
#define L_    4
#define DFF_  2048
#define NIN_  64
#define NPV_  16
#define NEMB_ 32
#define ROWS_ (B_ * T_)      // 4096
#define QKVW_ (3 * D_)       // 1536
#define KPAD_ 96             // padded input-proj K (enc: 96 exact, dec: 80 -> 96)
#define BROW_ 40             // LDS row stride in halves (32 data + 8 pad, bank-spread)

#if __has_builtin(__builtin_amdgcn_global_load_async_to_lds_b128) && \
    __has_builtin(__builtin_amdgcn_s_wait_asynccnt)
#define ASYNC_LDS 1
#else
#define ASYNC_LDS 0
#endif

typedef __attribute__((ext_vector_type(16))) _Float16 v16h;
typedef __attribute__((ext_vector_type(8)))  _Float16 v8h;
typedef __attribute__((ext_vector_type(8)))  float    v8f;
typedef int v4i __attribute__((vector_size(4 * sizeof(int))));

union HFrag { v16h v; v8h h[2]; };

__device__ __forceinline__ int imin(int a, int b) { return a < b ? a : b; }

// 16-bit A-matrix 16x32 (MxK) fragment: lanes 0-15 -> K[0..7]+K[16..23],
// lanes 16-31 -> K[8..15]+K[24..31] (ISA 7.12.2)
__device__ __forceinline__ v16h load_a_frag(const _Float16* __restrict__ A,
                                            int lda, int m, int k, int g) {
  HFrag f;
  const _Float16* p = A + (size_t)m * lda + k + 8 * g;
  f.h[0] = *(const v8h*)(p);
  f.h[1] = *(const v8h*)(p + 16);
  return f.v;
}
// 16-bit B-matrix 32x16 (KxN) fragment from LDS strip: lane n holds K[8g..8g+7]+K[16+8g..16+8g+7]
__device__ __forceinline__ v16h load_b_lds(const _Float16* lb, int nrow, int g) {
  HFrag f;
  const _Float16* p = lb + nrow * BROW_ + 8 * g;
  f.h[0] = *(const v8h*)(p);
  f.h[1] = *(const v8h*)(p + 16);
  return f.v;
}

template<int OUTF16, int RELU>
__device__ __forceinline__ void store_tile(void* __restrict__ C, size_t coff, int ldc,
    const float* __restrict__ bias, const float* __restrict__ R, int ldr,
    v8f acc, int rbase, int cbase, int M, int N, int g, int ln)
{
  int n = cbase + ln;
  if (n >= N) return;
  float bv = bias ? bias[n] : 0.0f;
  #pragma unroll
  for (int r = 0; r < 8; ++r) {
    int m = rbase + r + 8 * g;           // C/D layout: VGPR r -> row r + 8g
    if (m >= M) continue;
    float val = acc[r] + bv;
    if (RELU) val = fmaxf(val, 0.0f);
    size_t ci = coff + (size_t)m * ldc + n;
    if (OUTF16) {
      ((_Float16*)C)[ci] = (_Float16)val;
    } else {
      if (R) val += R[(size_t)m * ldr + n];
      ((float*)C)[ci] = val;
    }
  }
}

// Generic batched GEMM:  C[m,n] = sum_k A[m,k] * W[n,k]  (+bias, +residual, relu)
// Workgroup (8 waves) computes a 256x64 macro-tile: the 64x32 weight strip is
// cooperatively staged through LDS (async double-buffered when available);
// each wave computes 32 rows x 64 cols = 8 WMMA tiles per k-step.
// batch z decomposed as (bz, hz) with independent strides -> handles [B,H] attention.
template<int OUTF16, int RELU>
__global__ void __launch_bounds__(256) gemm_ab(
    const _Float16* __restrict__ A, const _Float16* __restrict__ W,
    void* __restrict__ C, const float* __restrict__ bias, const float* __restrict__ R,
    int M, int N, int K, int lda, int ldb, int ldc, int ldr,
    long long sAb, long long sAh, long long sBb, long long sBh,
    long long sCb, long long sCh, int Hdim, int nblocks)
{
  __shared__ _Float16 lds[2][64 * BROW_];

  int z = blockIdx.z;
  int bz = z / Hdim, hz = z - bz * Hdim;
  const _Float16* Ab = A + (size_t)(bz * sAb + hz * sAh);
  const _Float16* Wb = W + (size_t)(bz * sBb + hz * sBh);
  size_t coff = (size_t)(bz * sCb + hz * sCh);

  int bm = blockIdx.x / nblocks, bn = blockIdx.x - bm * nblocks;
  int col0 = bn << 6;                         // 64-wide block column strip
  int wv = threadIdx.x >> 5;
  int row0 = (bm << 8) + (wv << 5);           // 32 rows per wave
  int lane = threadIdx.x & 31;
  int g = lane >> 4, ln = lane & 15;
  int m0 = imin(row0 + ln, M - 1);
  int m1 = imin(row0 + 16 + ln, M - 1);

  // cooperative staging: thread tid covers (n_local = tid/4, 8-half chunk = tid%4)
  int tid = threadIdx.x;
  int snl = tid >> 2, sch = tid & 3;
  int srcn = imin(col0 + snl, N - 1);
  const _Float16* bsrc = Wb + (size_t)srcn * ldb + sch * 8;
  _Float16* bdst = &lds[0][0] + snl * BROW_ + sch * 8;

  auto stage = [&](int buf, int k) {
#if ASYNC_LDS
    __builtin_amdgcn_global_load_async_to_lds_b128(
        (__attribute__((address_space(1))) v4i*)(bsrc + k),
        (__attribute__((address_space(3))) v4i*)(bdst + buf * (64 * BROW_)),
        0, 0);
#else
    *(v8h*)(bdst + buf * (64 * BROW_)) = *(const v8h*)(bsrc + k);
#endif
  };

  v8f acc0[4] = {}, acc1[4] = {};
  stage(0, 0);
  int buf = 0;
  for (int k = 0; k < K; k += 32) {
    bool more = (k + 32) < K;                 // wave-uniform
    if (more) stage(buf ^ 1, k + 32);
#if ASYNC_LDS
    if (more) __builtin_amdgcn_s_wait_asynccnt(1);
    else      __builtin_amdgcn_s_wait_asynccnt(0);
#endif
    __syncthreads();
    v16h a0 = load_a_frag(Ab, lda, m0, k, g);
    v16h a1 = load_a_frag(Ab, lda, m1, k, g);
    if (more) {
      __builtin_prefetch(Ab + (size_t)m0 * lda + k + 32, 0, 1);   // global_prefetch_b8
      __builtin_prefetch(Ab + (size_t)m1 * lda + k + 32, 0, 1);
    }
    const _Float16* lb = &lds[0][0] + buf * (64 * BROW_);
    #pragma unroll
    for (int c = 0; c < 4; ++c) {
      v16h bf = load_b_lds(lb, c * 16 + ln, g);
      acc0[c] = __builtin_amdgcn_wmma_f32_16x16x32_f16(false, a0, false, bf, (short)0, acc0[c], false, false);
      acc1[c] = __builtin_amdgcn_wmma_f32_16x16x32_f16(false, a1, false, bf, (short)0, acc1[c], false, false);
    }
    __syncthreads();
    buf ^= 1;
  }
  #pragma unroll
  for (int c = 0; c < 4; ++c) {
    store_tile<OUTF16,RELU>(C, coff, ldc, bias, R, ldr, acc0[c], row0,      col0 + c * 16, M, N, g, ln);
    store_tile<OUTF16,RELU>(C, coff, ldc, bias, R, ldr, acc1[c], row0 + 16, col0 + c * 16, M, N, g, ln);
  }
}

// ---------------- LayerNorm: one wave per row of 512, f32 in -> f16 out --------
__global__ void __launch_bounds__(256) ln_rows(const float* __restrict__ x,
    const float* __restrict__ gam, const float* __restrict__ bet,
    _Float16* __restrict__ out, int rows)
{
  int wave = (blockIdx.x << 3) + (threadIdx.x >> 5);
  if (wave >= rows) return;
  int lane = threadIdx.x & 31;
  const float* xr = x + (size_t)wave * D_;
  float v[16];
  float s = 0.0f;
  #pragma unroll
  for (int i = 0; i < 16; ++i) { v[i] = xr[lane + 32 * i]; s += v[i]; }
  #pragma unroll
  for (int m = 16; m >= 1; m >>= 1) s += __shfl_xor(s, m, 32);
  float mean = s * (1.0f / D_);
  float q = 0.0f;
  #pragma unroll
  for (int i = 0; i < 16; ++i) { float d = v[i] - mean; q += d * d; }
  #pragma unroll
  for (int m = 16; m >= 1; m >>= 1) q += __shfl_xor(q, m, 32);
  float inv = rsqrtf(q * (1.0f / D_) + 1e-5f);
  _Float16* orow = out + (size_t)wave * D_;
  #pragma unroll
  for (int i = 0; i < 16; ++i) {
    int c = lane + 32 * i;
    orow[c] = (_Float16)((v[i] - mean) * inv * gam[c] + bet[c]);
  }
}

// ---------- softmax over rows of 256 (scale 1/8, optional causal), f32 -> f16 P ----
__global__ void __launch_bounds__(256) softmax_rows(const float* __restrict__ s,
    _Float16* __restrict__ p, int rows, int causal)
{
  int wave = (blockIdx.x << 3) + (threadIdx.x >> 5);
  if (wave >= rows) return;
  int lane = threadIdx.x & 31;
  int t = wave & (T_ - 1);                     // query position within sequence
  const float* sr = s + (size_t)wave * T_;
  float v[8];
  float mx = -3.0e38f;
  #pragma unroll
  for (int i = 0; i < 8; ++i) {
    int j = lane + 32 * i;
    float val = sr[j] * 0.125f;                // 1/sqrt(64)
    if (causal && j > t) val = -3.0e38f;
    v[i] = val;
    mx = fmaxf(mx, val);
  }
  #pragma unroll
  for (int m = 16; m >= 1; m >>= 1) mx = fmaxf(mx, __shfl_xor(mx, m, 32));
  float sum = 0.0f;
  #pragma unroll
  for (int i = 0; i < 8; ++i) { v[i] = expf(v[i] - mx); sum += v[i]; }
  #pragma unroll
  for (int m = 16; m >= 1; m >>= 1) sum += __shfl_xor(sum, m, 32);
  float inv = 1.0f / sum;
  _Float16* pr = p + (size_t)wave * T_;
  #pragma unroll
  for (int i = 0; i < 8; ++i) pr[lane + 32 * i] = (_Float16)(v[i] * inv);
}

// ------- transpose V: qkv[b,t, 2D + h*DH + d] -> vt[(b,h,d), t]  (f16) ----------
__global__ void transpose_v(const _Float16* __restrict__ qkv, _Float16* __restrict__ vt)
{
  int idx = blockIdx.x * blockDim.x + threadIdx.x;
  int total = B_ * H_ * DH_ * T_;
  if (idx >= total) return;
  int t = idx % T_; int r = idx / T_;
  int d = r % DH_;  int bh = r / DH_;
  int h = bh % H_;  int b = bh / H_;
  vt[idx] = qkv[((size_t)(b * T_ + t)) * QKVW_ + 2 * D_ + h * DH_ + d];
}

// ---------------- encoder input concat: [x_cv ; scen_emb[scenario]] -> f16 ------
__global__ void concat_enc(const float* __restrict__ x_cv, const float* __restrict__ scen_emb,
                           const int* __restrict__ scenario, _Float16* __restrict__ out)
{
  int idx = blockIdx.x * blockDim.x + threadIdx.x;
  int total = ROWS_ * KPAD_;
  if (idx >= total) return;
  int c = idx % KPAD_; int bt = idx / KPAD_; int b = bt / T_;
  float v;
  if (c < NIN_) v = x_cv[(size_t)bt * NIN_ + c];
  else          v = scen_emb[scenario[b] * NEMB_ + (c - NIN_)];
  out[idx] = (_Float16)v;
}

// ---------- decoder input concat: [x_cv_target ; pv_in ; zero-pad] -> f16 -------
__global__ void concat_dec(const float* __restrict__ x_tgt, const float* __restrict__ pv_init,
                           const float* __restrict__ pv_teacher, _Float16* __restrict__ out)
{
  int idx = blockIdx.x * blockDim.x + threadIdx.x;
  int total = ROWS_ * KPAD_;
  if (idx >= total) return;
  int c = idx % KPAD_; int bt = idx / KPAD_;
  int b = bt / T_; int t = bt % T_;
  float v = 0.0f;
  if (c < NIN_) v = x_tgt[(size_t)bt * NIN_ + c];
  else if (c < NIN_ + NPV_) {
    int cc = c - NIN_;
    v = (t == 0) ? pv_init[b * NPV_ + cc]
                 : pv_teacher[((size_t)b * T_ + (t - 1)) * NPV_ + cc];
  }
  out[idx] = (_Float16)v;
}

// ---------------- add sinusoidal positional encoding to f32 [4096,512] ----------
__global__ void add_pe(float* __restrict__ x)
{
  int idx = blockIdx.x * blockDim.x + threadIdx.x;
  int total = ROWS_ * D_;
  if (idx >= total) return;
  int n = idx % D_; int m = idx / D_; int t = m % T_;
  int i2 = n & ~1;
  float div = expf(-(float)i2 * (9.2103403719761836f / (float)D_));
  float ang = (float)t * div;
  x[idx] += (n & 1) ? cosf(ang) : sinf(ang);
}

// ---------------- elementwise f32 -> f16 ----------------------------------------
__global__ void cvt_f16(const float* __restrict__ src, _Float16* __restrict__ dst, int n)
{
  int idx = blockIdx.x * blockDim.x + threadIdx.x;
  if (idx < n) dst[idx] = (_Float16)src[idx];
}

// ------------ weight convert (+ optional K pad): f32[rows,srcK]->f16[rows,dstK] --
__global__ void w2h(const float* __restrict__ src, _Float16* __restrict__ dst,
                    int total, int srcK, int dstK)
{
  int idx = blockIdx.x * blockDim.x + threadIdx.x;
  if (idx >= total) return;
  int c = idx % dstK; int r = idx / dstK;
  dst[idx] = (c < srcK) ? (_Float16)src[(size_t)r * srcK + c] : (_Float16)0.0f;
}

// =================================================================================
extern "C" void kernel_launch(void* const* d_in, const int* in_sizes, int n_in,
                              void* d_out, int out_size, void* d_ws, size_t ws_size,
                              hipStream_t stream)
{
  (void)in_sizes; (void)n_in; (void)out_size;
  // ---- input pointers (setup_inputs() dict order; nested dicts in insertion order)
  const float* x_cv        = (const float*)d_in[0];
  const float* x_cv_target = (const float*)d_in[1];
  const float* pv_init     = (const float*)d_in[2];
  const float* pv_teacher  = (const float*)d_in[3];
  const float* scen_emb    = (const float*)d_in[4];
  const float* enc_proj_w  = (const float*)d_in[5];
  const float* enc_proj_b  = (const float*)d_in[6];
  const float* dec_proj_w  = (const float*)d_in[7];
  const float* dec_proj_b  = (const float*)d_in[8];
  // encoder params (9..20): ln1_g ln1_b sa_qkv_w sa_qkv_b sa_out_w sa_out_b ln2_g ln2_b ff1_w ff1_b ff2_w ff2_b
  const float* e_ln1_g  = (const float*)d_in[9];
  const float* e_ln1_b  = (const float*)d_in[10];
  const float* e_qkv_w  = (const float*)d_in[11];
  const float* e_qkv_b  = (const float*)d_in[12];
  const float* e_out_w  = (const float*)d_in[13];
  const float* e_out_b  = (const float*)d_in[14];
  const float* e_ln2_g  = (const float*)d_in[15];
  const float* e_ln2_b  = (const float*)d_in[16];
  const float* e_ff1_w  = (const float*)d_in[17];
  const float* e_ff1_b  = (const float*)d_in[18];
  const float* e_ff2_w  = (const float*)d_in[19];
  const float* e_ff2_b  = (const float*)d_in[20];
  // decoder params (21..38): base 12 + ca_qkv_w ca_qkv_b ca_out_w ca_out_b ln3_g ln3_b
  const float* d_ln1_g  = (const float*)d_in[21];
  const float* d_ln1_b  = (const float*)d_in[22];
  const float* d_qkv_w  = (const float*)d_in[23];
  const float* d_qkv_b  = (const float*)d_in[24];
  const float* d_out_w  = (const float*)d_in[25];
  const float* d_out_b  = (const float*)d_in[26];
  const float* d_ln2_g  = (const float*)d_in[27];
  const float* d_ln2_b  = (const float*)d_in[28];
  const float* d_ff1_w  = (const float*)d_in[29];
  const float* d_ff1_b  = (const float*)d_in[30];
  const float* d_ff2_w  = (const float*)d_in[31];
  const float* d_ff2_b  = (const float*)d_in[32];
  const float* d_caqkv_w = (const float*)d_in[33];
  const float* d_caqkv_b = (const float*)d_in[34];
  const float* d_caout_w = (const float*)d_in[35];
  const float* d_caout_b = (const float*)d_in[36];
  const float* d_ln3_g  = (const float*)d_in[37];
  const float* d_ln3_b  = (const float*)d_in[38];
  const float* fc1_w = (const float*)d_in[39];
  const float* fc1_b = (const float*)d_in[40];
  const float* fc2_w = (const float*)d_in[41];
  const float* fc2_b = (const float*)d_in[42];
  const int*   scenario = (const int*)d_in[43];
  float* outp = (float*)d_out;

  // ---- workspace bump allocator -------------------------------------------------
  char* base = (char*)d_ws;
  size_t off = 0;
  auto alloc = [&](size_t bytes) -> void* {
    void* p = base + off;
    off += (bytes + 255) & ~(size_t)255;
    return p;
  };
  auto allocH = [&](size_t elems) { return (_Float16*)alloc(elems * sizeof(_Float16)); };
  auto allocF = [&](size_t elems) { return (float*)alloc(elems * sizeof(float)); };

  // f16 weight copies
  _Float16* encW  = allocH((size_t)D_ * KPAD_);
  _Float16* decW  = allocH((size_t)D_ * KPAD_);
  _Float16* eQKV  = allocH((size_t)L_ * QKVW_ * D_);
  _Float16* eOUT  = allocH((size_t)L_ * D_ * D_);
  _Float16* eFF1  = allocH((size_t)L_ * DFF_ * D_);
  _Float16* eFF2  = allocH((size_t)L_ * D_ * DFF_);
  _Float16* dQKV  = allocH((size_t)L_ * QKVW_ * D_);
  _Float16* dOUT  = allocH((size_t)L_ * D_ * D_);
  _Float16* dCAQ  = allocH((size_t)L_ * QKVW_ * D_);
  _Float16* dCAO  = allocH((size_t)L_ * D_ * D_);
  _Float16* dFF1  = allocH((size_t)L_ * DFF_ * D_);
  _Float16* dFF2  = allocH((size_t)L_ * D_ * DFF_);
  _Float16* FC1   = allocH((size_t)128 * D_);
  _Float16* FC2   = allocH((size_t)NPV_ * 128);
  // activations
  _Float16* cin   = allocH((size_t)ROWS_ * KPAD_);     // concat input (enc/dec)
  float*    x     = allocF((size_t)ROWS_ * D_);        // running state (f32)
  _Float16* xn    = allocH((size_t)ROWS_ * D_);        // LN output / head input (f16)
  _Float16* qkv   = allocH((size_t)ROWS_ * QKVW_);     // q|k|v packed
  _Float16* vt    = allocH((size_t)B_ * H_ * DH_ * T_);
  float*    sc    = allocF((size_t)B_ * H_ * T_ * T_); // attention scores
  _Float16* P     = allocH((size_t)B_ * H_ * T_ * T_); // softmax probs
  _Float16* aO    = allocH((size_t)ROWS_ * D_);        // attention output (heads merged)
  _Float16* ffh   = allocH((size_t)ROWS_ * DFF_);      // FF hidden / head hidden
  _Float16* mem16 = allocH((size_t)ROWS_ * D_);        // encoder memory (f16)
  if (off > ws_size) return;                           // workspace too small

  // ---- small launch helpers -----------------------------------------------------
  auto ew = [&](int total) { return dim3((total + 255) / 256); };
  auto cvtW = [&](const float* src, _Float16* dst, int rows, int srcK, int dstK) {
    int total = rows * dstK;
    w2h<<<ew(total), 256, 0, stream>>>(src, dst, total, srcK, dstK);
  };
  auto gemm = [&](int outf16, int relu,
                  const _Float16* A, const _Float16* W, void* C,
                  const float* bias, const float* R,
                  int M, int N, int K, int lda, int ldb, int ldc, int ldr,
                  long long sAb, long long sAh, long long sBb, long long sBh,
                  long long sCb, long long sCh, int Hd, int batch) {
    int mb = (M + 255) / 256, nb = (N + 63) / 64;
    dim3 grid(mb * nb, 1, batch), block(256);
    if (outf16) {
      if (relu) gemm_ab<1,1><<<grid, block, 0, stream>>>(A, W, C, bias, R, M, N, K,
          lda, ldb, ldc, ldr, sAb, sAh, sBb, sBh, sCb, sCh, Hd, nb);
      else      gemm_ab<1,0><<<grid, block, 0, stream>>>(A, W, C, bias, R, M, N, K,
          lda, ldb, ldc, ldr, sAb, sAh, sBb, sBh, sCb, sCh, Hd, nb);
    } else      gemm_ab<0,0><<<grid, block, 0, stream>>>(A, W, C, bias, R, M, N, K,
          lda, ldb, ldc, ldr, sAb, sAh, sBb, sBh, sCb, sCh, Hd, nb);
  };
  auto ln = [&](const float* in, const float* g, const float* b, _Float16* o) {
    ln_rows<<<(ROWS_ + 7) / 8, 256, 0, stream>>>(in, g, b, o, ROWS_);
  };
  // full attention: qkv (f16, packed q|k|v, head-interleaved) -> aO
  auto attention = [&](int causal) {
    int tv = B_ * H_ * DH_ * T_;
    transpose_v<<<ew(tv), 256, 0, stream>>>(qkv, vt);
    // scores[b,h] = Q @ K^T   (M=N=T, K=DH)
    gemm(0, 0, qkv, qkv + D_, sc, nullptr, nullptr,
         T_, T_, DH_, QKVW_, QKVW_, T_, 0,
         (long long)T_ * QKVW_, DH_, (long long)T_ * QKVW_, DH_,
         (long long)H_ * T_ * T_, (long long)T_ * T_, H_, B_ * H_);
    int rowsS = B_ * H_ * T_;
    softmax_rows<<<(rowsS + 7) / 8, 256, 0, stream>>>(sc, P, rowsS, causal);
    // aO[b,h] = P @ V   (via V^T as the [N,K] operand)
    gemm(1, 0, P, vt, aO, nullptr, nullptr,
         T_, DH_, T_, T_, T_, D_, 0,
         (long long)H_ * T_ * T_, (long long)T_ * T_,
         (long long)H_ * DH_ * T_, (long long)DH_ * T_,
         (long long)T_ * D_, DH_, H_, B_ * H_);
  };

  // ---- convert weights to f16 ----------------------------------------------------
  cvtW(enc_proj_w, encW, D_, KPAD_, KPAD_);            // 96 -> 96
  cvtW(dec_proj_w, decW, D_, NIN_ + NPV_, KPAD_);      // 80 -> 96 zero-padded
  cvtW(e_qkv_w, eQKV, L_ * QKVW_, D_, D_);
  cvtW(e_out_w, eOUT, L_ * D_, D_, D_);
  cvtW(e_ff1_w, eFF1, L_ * DFF_, D_, D_);
  cvtW(e_ff2_w, eFF2, L_ * D_, DFF_, DFF_);
  cvtW(d_qkv_w, dQKV, L_ * QKVW_, D_, D_);
  cvtW(d_out_w, dOUT, L_ * D_, D_, D_);
  cvtW(d_caqkv_w, dCAQ, L_ * QKVW_, D_, D_);
  cvtW(d_caout_w, dCAO, L_ * D_, D_, D_);
  cvtW(d_ff1_w, dFF1, L_ * DFF_, D_, D_);
  cvtW(d_ff2_w, dFF2, L_ * D_, DFF_, DFF_);
  cvtW(fc1_w, FC1, 128, D_, D_);
  cvtW(fc2_w, FC2, NPV_, 128, 128);

  // ================================ ENCODER =======================================
  concat_enc<<<ew(ROWS_ * KPAD_), 256, 0, stream>>>(x_cv, scen_emb, scenario, cin);
  gemm(0, 0, cin, encW, x, enc_proj_b, nullptr,
       ROWS_, D_, KPAD_, KPAD_, KPAD_, D_, 0, 0, 0, 0, 0, 0, 0, 1, 1);
  add_pe<<<ew(ROWS_ * D_), 256, 0, stream>>>(x);

  for (int i = 0; i < L_; ++i) {
    ln(x, e_ln1_g + i * D_, e_ln1_b + i * D_, xn);
    gemm(1, 0, xn, eQKV + (size_t)i * QKVW_ * D_, qkv, e_qkv_b + i * QKVW_, nullptr,
         ROWS_, QKVW_, D_, D_, D_, QKVW_, 0, 0, 0, 0, 0, 0, 0, 1, 1);
    attention(/*causal=*/0);
    gemm(0, 0, aO, eOUT + (size_t)i * D_ * D_, x, e_out_b + i * D_, x,
         ROWS_, D_, D_, D_, D_, D_, D_, 0, 0, 0, 0, 0, 0, 1, 1);
    ln(x, e_ln2_g + i * D_, e_ln2_b + i * D_, xn);
    gemm(1, 1, xn, eFF1 + (size_t)i * DFF_ * D_, ffh, e_ff1_b + i * DFF_, nullptr,
         ROWS_, DFF_, D_, D_, D_, DFF_, 0, 0, 0, 0, 0, 0, 0, 1, 1);
    gemm(0, 0, ffh, eFF2 + (size_t)i * D_ * DFF_, x, e_ff2_b + i * D_, x,
         ROWS_, D_, DFF_, DFF_, DFF_, D_, D_, 0, 0, 0, 0, 0, 0, 1, 1);
  }
  cvt_f16<<<ew(ROWS_ * D_), 256, 0, stream>>>(x, mem16, ROWS_ * D_);

  // ================================ DECODER =======================================
  concat_dec<<<ew(ROWS_ * KPAD_), 256, 0, stream>>>(x_cv_target, pv_init, pv_teacher, cin);
  gemm(0, 0, cin, decW, x, dec_proj_b, nullptr,
       ROWS_, D_, KPAD_, KPAD_, KPAD_, D_, 0, 0, 0, 0, 0, 0, 0, 1, 1);
  add_pe<<<ew(ROWS_ * D_), 256, 0, stream>>>(x);

  for (int i = 0; i < L_; ++i) {
    // --- causal self-attention
    ln(x, d_ln1_g + i * D_, d_ln1_b + i * D_, xn);
    gemm(1, 0, xn, dQKV + (size_t)i * QKVW_ * D_, qkv, d_qkv_b + i * QKVW_, nullptr,
         ROWS_, QKVW_, D_, D_, D_, QKVW_, 0, 0, 0, 0, 0, 0, 0, 1, 1);
    attention(/*causal=*/1);
    gemm(0, 0, aO, dOUT + (size_t)i * D_ * D_, x, d_out_b + i * D_, x,
         ROWS_, D_, D_, D_, D_, D_, D_, 0, 0, 0, 0, 0, 0, 1, 1);
    // --- cross-attention (q from x, k/v from encoder memory)
    ln(x, d_ln2_g + i * D_, d_ln2_b + i * D_, xn);
    gemm(1, 0, xn, dCAQ + (size_t)i * QKVW_ * D_, qkv, d_caqkv_b + i * QKVW_, nullptr,
         ROWS_, D_, D_, D_, D_, QKVW_, 0, 0, 0, 0, 0, 0, 0, 1, 1);
    gemm(1, 0, mem16, dCAQ + (size_t)i * QKVW_ * D_ + (size_t)D_ * D_, qkv + D_,
         d_caqkv_b + i * QKVW_ + D_, nullptr,
         ROWS_, 2 * D_, D_, D_, D_, QKVW_, 0, 0, 0, 0, 0, 0, 0, 1, 1);
    attention(/*causal=*/0);
    gemm(0, 0, aO, dCAO + (size_t)i * D_ * D_, x, d_caout_b + i * D_, x,
         ROWS_, D_, D_, D_, D_, D_, D_, 0, 0, 0, 0, 0, 0, 1, 1);
    // --- feed-forward
    ln(x, d_ln3_g + i * D_, d_ln3_b + i * D_, xn);
    gemm(1, 1, xn, dFF1 + (size_t)i * DFF_ * D_, ffh, d_ff1_b + i * DFF_, nullptr,
         ROWS_, DFF_, D_, D_, D_, DFF_, 0, 0, 0, 0, 0, 0, 0, 1, 1);
    gemm(0, 0, ffh, dFF2 + (size_t)i * D_ * DFF_, x, d_ff2_b + i * D_, x,
         ROWS_, D_, DFF_, DFF_, DFF_, D_, D_, 0, 0, 0, 0, 0, 0, 1, 1);
  }

  // ================================ HEAD ==========================================
  cvt_f16<<<ew(ROWS_ * D_), 256, 0, stream>>>(x, xn, ROWS_ * D_);
  gemm(1, 1, xn, FC1, ffh, fc1_b, nullptr,
       ROWS_, 128, D_, D_, D_, 128, 0, 0, 0, 0, 0, 0, 0, 1, 1);
  gemm(0, 0, ffh, FC2, outp, fc2_b, nullptr,
       ROWS_, NPV_, 128, 128, 128, NPV_, 0, 0, 0, 0, 0, 0, 0, 1, 1);
}
